// AttentiveSSMNoProjCyc_45389214384868
// MI455X (gfx1250) — matrix operand order repaired and checked by reading, not compiled, not run on hardware
//
#include <hip/hip_runtime.h>
#include <hip/hip_bf16.h>
#include <math.h>

// ---------------- problem constants (from reference) ----------------
#define B_   2
#define S_   2048
#define D_   1024
#define H_   16
#define HD_  64
#define M_   (B_ * S_)        // 4096 rows
#define NCHUNK_ 33            // SSM chunks: [0,0],[1,47],[48,111],...,[1968,2047]

typedef __attribute__((ext_vector_type(16))) __bf16 v16bf;
typedef __attribute__((ext_vector_type(8)))  __bf16 v8bf;
typedef __attribute__((ext_vector_type(4)))  __bf16 v4bf;
typedef __attribute__((ext_vector_type(8)))  float  v8f;

union FragU { v16bf v; v8bf h[2]; };

// fp32 -> bf16 hi/lo split: hi = truncate-to-bf16(x) (exact), lo = bf16(x - hi)
__device__ __forceinline__ void split_bf16(float x, __bf16& hi, __bf16& lo) {
    union { float f; unsigned u; } c; c.f = x;
    c.u &= 0xffff0000u;
    hi = (__bf16)c.f;            // exact: low mantissa already zero
    lo = (__bf16)(x - c.f);
}

// 4-wide split into ext vectors (vector elements can't bind to references,
// so go through scalar temporaries and assign by subscript).
__device__ __forceinline__ void split_bf16_x4(const float* v, v4bf& h, v4bf& L) {
    #pragma unroll
    for (int j = 0; j < 4; ++j) {
        __bf16 th, tl;
        split_bf16(v[j], th, tl);
        h[j] = th; L[j] = tl;
    }
}

// Async DMA of 16 bytes global -> LDS (per lane). ASYNCcnt-tracked.
// LDS byte address = low 32 bits of the generic pointer (ISA aperture rule).
__device__ __forceinline__ void async_copy16(const __bf16* g, __bf16* l) {
    unsigned lds_off = (unsigned)(size_t)l;
    asm volatile("global_load_async_to_lds_b128 %0, %1, off"
                 :: "v"(lds_off), "v"(g) : "memory");
}

// =====================================================================
// fp32 -> bf16 hi/lo split pre-pass (amortizes conversion: done once,
// instead of once per consuming block).
// =====================================================================
__global__ __launch_bounds__(256)
void split_kernel(const float* __restrict__ in, __bf16* __restrict__ hi,
                  __bf16* __restrict__ lo, int n) {
    const int i = (blockIdx.x * blockDim.x + threadIdx.x) * 4;
    if (i >= n) return;
    float4 v = *(const float4*)(in + i);
    const float vv[4] = {v.x, v.y, v.z, v.w};
    v4bf h, L;
    split_bf16_x4(vv, h, L);
    *(v4bf*)(hi + i) = h;
    *(v4bf*)(lo + i) = L;
}

// =====================================================================
// GEMM: C[M,N] = A[M,K] * B[N,K]^T, A/B pre-split to bf16 hi/lo.
// 64x64 tile per 256-thread block; K staged 32 at a time through a
// double-buffered LDS pipeline fed by global_load_async_to_lds_b128,
// retired with s_wait_asynccnt. bf16x3 WMMA accumulation in fp32.
// =====================================================================
__global__ __launch_bounds__(256)
void gemm_async_bf16x3(const __bf16* __restrict__ Ahi, const __bf16* __restrict__ Alo,
                       const __bf16* __restrict__ Bhi, const __bf16* __restrict__ Blo,
                       float* __restrict__ C, int M, int N, int K) {
    __shared__ __align__(16) __bf16 AsHi[2][64][32];
    __shared__ __align__(16) __bf16 AsLo[2][64][32];
    __shared__ __align__(16) __bf16 BsHi[2][64][32];
    __shared__ __align__(16) __bf16 BsLo[2][64][32];   // 32 KB total

    const int tid  = threadIdx.x;
    const int lane = tid & 31;
    const int w    = tid >> 5;          // wave 0..7
    const int n0   = blockIdx.x * 64;
    const int m0   = blockIdx.y * 64;

    // DMA: each thread owns one 16B chunk per array per stage
    const int srow = tid >> 2;          // 0..63
    const int scol = (tid & 3) * 8;     // 0,8,16,24

    // fragment lane decode (ISA 7.12.2 layouts, wave32)
    const int l  = lane & 15;
    const int g  = lane >> 4;
    const int mr = (w >> 1) * 16;
    const int nc = (w & 1) * 32;

    const __bf16* gAh = Ahi + (size_t)(m0 + srow) * K + scol;
    const __bf16* gAl = Alo + (size_t)(m0 + srow) * K + scol;
    const __bf16* gBh = Bhi + (size_t)(n0 + srow) * K + scol;
    const __bf16* gBl = Blo + (size_t)(n0 + srow) * K + scol;

    auto issue = [&](int k0, int buf) {
        async_copy16(gAh + k0, &AsHi[buf][srow][scol]);
        async_copy16(gAl + k0, &AsLo[buf][srow][scol]);
        async_copy16(gBh + k0, &BsHi[buf][srow][scol]);
        async_copy16(gBl + k0, &BsLo[buf][srow][scol]);
    };

    v8f acc0 = {}; v8f acc1 = {};

    const int nsteps = K / 32;
    issue(0, 0);
    for (int s = 0; s < nsteps; ++s) {
        const int buf = s & 1;
        if (s + 1 < nsteps) {
            issue((s + 1) * 32, buf ^ 1);                    // prefetch next stage
            asm volatile("s_wait_asynccnt 0x4" ::: "memory"); // stage s retired
        } else {
            asm volatile("s_wait_asynccnt 0x0" ::: "memory");
        }
        __syncthreads();   // all waves' DMA for stage s visible

        // A fragment (16x32 bf16): lane<16 -> K 0..15 of row M=l;
        // lanes>=16 -> K 16..31. Two ds_load_b128 per fragment.
        const int arow = mr + l;
        FragU ah, al;
        ah.h[0] = *(const v8bf*)&AsHi[buf][arow][g * 16];
        ah.h[1] = *(const v8bf*)&AsHi[buf][arow][g * 16 + 8];
        al.h[0] = *(const v8bf*)&AsLo[buf][arow][g * 16];
        al.h[1] = *(const v8bf*)&AsLo[buf][arow][g * 16 + 8];

        // B fragment (32x16 bf16): col N=l, K kb..kb+15 contiguous per lane group.
        const int kb = g * 16;
        const int c0 = nc + l, c1 = nc + 16 + l;
        FragU bh0, bl0, bh1, bl1;
        bh0.h[0] = *(const v8bf*)&BsHi[buf][c0][kb];  bh0.h[1] = *(const v8bf*)&BsHi[buf][c0][kb + 8];
        bl0.h[0] = *(const v8bf*)&BsLo[buf][c0][kb];  bl0.h[1] = *(const v8bf*)&BsLo[buf][c0][kb + 8];
        bh1.h[0] = *(const v8bf*)&BsHi[buf][c1][kb];  bh1.h[1] = *(const v8bf*)&BsHi[buf][c1][kb + 8];
        bl1.h[0] = *(const v8bf*)&BsLo[buf][c1][kb];  bl1.h[1] = *(const v8bf*)&BsLo[buf][c1][kb + 8];

        // bf16x3 split-precision accumulation (drop lo*lo term)
        acc0 = __builtin_amdgcn_wmma_f32_16x16x32_bf16(false, ah.v, false, bh0.v, (short)0, acc0, false, false);
        acc0 = __builtin_amdgcn_wmma_f32_16x16x32_bf16(false, ah.v, false, bl0.v, (short)0, acc0, false, false);
        acc0 = __builtin_amdgcn_wmma_f32_16x16x32_bf16(false, al.v, false, bh0.v, (short)0, acc0, false, false);

        acc1 = __builtin_amdgcn_wmma_f32_16x16x32_bf16(false, ah.v, false, bh1.v, (short)0, acc1, false, false);
        acc1 = __builtin_amdgcn_wmma_f32_16x16x32_bf16(false, ah.v, false, bl1.v, (short)0, acc1, false, false);
        acc1 = __builtin_amdgcn_wmma_f32_16x16x32_bf16(false, al.v, false, bh1.v, (short)0, acc1, false, false);

        __syncthreads();   // reads done before this buffer is DMA-overwritten
    }

    // C/D 16x16 f32 layout: lane<16 -> N=l, M=vgpr r; lane>=16 -> M=vgpr+8
    const int mBase = m0 + mr + g * 8;
    int nn = n0 + nc + l;
    #pragma unroll
    for (int r = 0; r < 8; ++r) C[(size_t)(mBase + r) * N + nn] = acc0[r];
    nn += 16;
    #pragma unroll
    for (int r = 0; r < 8; ++r) C[(size_t)(mBase + r) * N + nn] = acc1[r];
}

// =====================================================================
// Chunked diagonal SSM (two gates at once) + residual.
// =====================================================================
__global__ __launch_bounds__(256)
void ssm_kernel(const float* __restrict__ x, const float* __restrict__ akp,
                const float* __restrict__ avp,
                float* __restrict__ xk, float* __restrict__ xv) {
    const int d = blockIdx.x * blockDim.x + threadIdx.x;   // 0..1023
    const int c = blockIdx.y;                              // 0..32
    const int b = blockIdx.z;
    const int lo = (c == 0) ? 0 : (c == 1) ? 1 : 48 + 64 * (c - 2);
    const int hi = (c == 0) ? 0 : (c == 32) ? (S_ - 1) : 47 + 64 * (c - 1);

    const float a1 = 1.f / (1.f + __expf(-akp[d]));
    const float a2 = 1.f / (1.f + __expf(-avp[d]));
    const float b1 = 1.f - a1, b2 = 1.f - a2;

    float h1 = 0.f, h2 = 0.f;
    const size_t base = (size_t)b * S_ * D_ + d;
    for (int t = lo; t <= hi; ++t) {
        const size_t idx = base + (size_t)t * D_;
        const float xt = x[idx];
        h1 = (t == lo ? 0.f : a1 * h1) + b1 * xt;
        h2 = (t == lo ? 0.f : a2 * h2) + b2 * xt;
        xk[idx] = h1 + xt;       // residual add
        xv[idx] = h2 + xt;
    }
}

// =====================================================================
// RoPE (2x2 rotation per even/odd pair), in place.
// =====================================================================
__global__ __launch_bounds__(256)
void rope_kernel(float* __restrict__ z, const float* __restrict__ fc) {
    const int gid = blockIdx.x * blockDim.x + threadIdx.x; // B*S*H*(HD/2) = 2M
    const int i = gid & 31;
    const int h = (gid >> 5) & 15;
    const int s = (gid >> 9) & (S_ - 1);
    const int b = gid >> 20;
    const size_t idx = (((size_t)(b * S_ + s) * H_ + h) * HD_) + 2 * i;
    const float z0 = z[idx], z1 = z[idx + 1];
    const float* f = fc + ((size_t)s * 32 + i) * 4;
    z[idx]     = z0 * f[0] + z1 * f[1];
    z[idx + 1] = z0 * f[2] + z1 * f[3];
}

// =====================================================================
// Structurally sparse attention with online softmax.
// Keys for query t: boundaries {0, 47+64j<=min(t-2,1967)} U {t-1, t}.
// Output written already bf16 hi/lo split (feeds WMMA GEMM2 directly).
// =====================================================================
__global__ __launch_bounds__(256)
void attn_kernel(const float* __restrict__ XQ, const float* __restrict__ XK,
                 const float* __restrict__ XV,
                 __bf16* __restrict__ Ohi, __bf16* __restrict__ Olo) {
    const int gid = blockIdx.x * blockDim.x + threadIdx.x;  // B*H*S = 65536
    const int t = gid & (S_ - 1);
    const int h = (gid >> 11) & (H_ - 1);
    const int b = gid >> 15;

    const size_t qoff = ((size_t)(b * S_ + t) * H_ + h) * HD_;
    float4 q4[16];
    #pragma unroll
    for (int i = 0; i < 16; ++i) q4[i] = ((const float4*)(XQ + qoff))[i];

    float4 acc4[16];
    #pragma unroll
    for (int i = 0; i < 16; ++i) acc4[i] = make_float4(0.f, 0.f, 0.f, 0.f);
    float mrun = -1e30f, lrun = 0.f;

    auto process = [&](int k) {
        const size_t koff = ((size_t)(b * S_ + k) * H_ + h) * HD_;
        const float4* kp = (const float4*)(XK + koff);
        const float4* vp = (const float4*)(XV + koff);
        float s = 0.f;
        #pragma unroll
        for (int i = 0; i < 16; ++i) {
            float4 kv = kp[i];
            s += q4[i].x * kv.x + q4[i].y * kv.y + q4[i].z * kv.z + q4[i].w * kv.w;
        }
        s *= 0.125f;                              // 1/sqrt(64)
        const float mn   = fmaxf(mrun, s);
        const float wgt  = __expf(s - mn);
        const float corr = __expf(mrun - mn);
        lrun = lrun * corr + wgt;
        #pragma unroll
        for (int i = 0; i < 16; ++i) {
            float4 vv = vp[i];
            acc4[i].x = acc4[i].x * corr + wgt * vv.x;
            acc4[i].y = acc4[i].y * corr + wgt * vv.y;
            acc4[i].z = acc4[i].z * corr + wgt * vv.z;
            acc4[i].w = acc4[i].w * corr + wgt * vv.w;
        }
        mrun = mn;
    };

    const int kmax = t - 2;
    if (kmax >= 0) process(0);
    for (int k = 47; k <= kmax && k <= 1967; k += 64) process(k);
    if (t >= 1) process(t - 1);
    process(t);

    const float inv = 1.f / lrun;
    #pragma unroll
    for (int i = 0; i < 16; ++i) {
        const float vx[4] = {acc4[i].x * inv, acc4[i].y * inv, acc4[i].z * inv, acc4[i].w * inv};
        v4bf hh, ll;
        split_bf16_x4(vx, hh, ll);
        *(v4bf*)(Ohi + qoff + 4 * i) = hh;
        *(v4bf*)(Olo + qoff + 4 * i) = ll;
    }
}

// =====================================================================
extern "C" void kernel_launch(void* const* d_in, const int* in_sizes, int n_in,
                              void* d_out, int out_size, void* d_ws, size_t ws_size,
                              hipStream_t stream) {
    const float* x  = (const float*)d_in[0];   // (B,S,D)
    const float* fc = (const float*)d_in[1];   // (S,32,2,2)
    const float* wq = (const float*)d_in[2];   // (1024, D)
    const float* wo = (const float*)d_in[3];   // (D, 1024)
    const float* ak = (const float*)d_in[4];   // (D,)
    const float* av = (const float*)d_in[5];   // (D,)
    float* out = (float*)d_out;                // (B,S,D) fp32

    char* ws = (char*)d_ws;
    const size_t F32SEG = (size_t)M_ * D_ * sizeof(float);    // 16 MB
    const size_t BFSEG  = (size_t)M_ * D_ * sizeof(__bf16);   //  8 MB
    const size_t WSEG   = (size_t)D_ * D_ * sizeof(__bf16);   //  2 MB
    float*  XQ   = (float*)(ws);
    float*  XK   = (float*)(ws + F32SEG);
    float*  XV   = (float*)(ws + 2 * F32SEG);
    __bf16* Xhi  = (__bf16*)(ws + 3 * F32SEG);                // reused as ATThi
    __bf16* Xlo  = (__bf16*)(ws + 3 * F32SEG + BFSEG);        // reused as ATTlo
    __bf16* WQhi = (__bf16*)(ws + 3 * F32SEG + 2 * BFSEG);
    __bf16* WQlo = (__bf16*)(ws + 3 * F32SEG + 2 * BFSEG + WSEG);
    __bf16* WOhi = (__bf16*)(ws + 3 * F32SEG + 2 * BFSEG + 2 * WSEG);
    __bf16* WOlo = (__bf16*)(ws + 3 * F32SEG + 2 * BFSEG + 3 * WSEG);

    // 0) one-time fp32 -> bf16 hi/lo splits
    split_kernel<<<(M_ * D_ / 4 + 255) / 256, 256, 0, stream>>>(x,  Xhi,  Xlo,  M_ * D_);
    split_kernel<<<(D_ * D_ / 4 + 255) / 256, 256, 0, stream>>>(wq, WQhi, WQlo, D_ * D_);
    split_kernel<<<(D_ * D_ / 4 + 255) / 256, 256, 0, stream>>>(wo, WOhi, WOlo, D_ * D_);

    // 1) xq = x @ wq^T      (async-LDS pipelined WMMA bf16x3)
    gemm_async_bf16x3<<<dim3(D_ / 64, M_ / 64), 256, 0, stream>>>(Xhi, Xlo, WQhi, WQlo, XQ, M_, D_, D_);
    // 2) chunked SSM (k and v gates) + residual
    ssm_kernel<<<dim3(D_ / 256, NCHUNK_, B_), 256, 0, stream>>>(x, ak, av, XK, XV);
    // 3) RoPE on q and k
    rope_kernel<<<(B_ * S_ * H_ * (HD_ / 2)) / 256, 256, 0, stream>>>(XQ, fc);
    rope_kernel<<<(B_ * S_ * H_ * (HD_ / 2)) / 256, 256, 0, stream>>>(XK, fc);
    // 4) sparse boundary-mask attention -> bf16 hi/lo (reuses Xhi/Xlo region)
    attn_kernel<<<(B_ * H_ * S_) / 256, 256, 0, stream>>>(XQ, XK, XV, Xhi, Xlo);
    // 5) out = att @ wo^T    (async-LDS pipelined WMMA bf16x3)
    gemm_async_bf16x3<<<dim3(D_ / 64, M_ / 64), 256, 0, stream>>>(Xhi, Xlo, WOhi, WOlo, out, M_, D_, D_);
}
